// MultiHeadedAttention_14611478741370
// MI455X (gfx1250) — matrix operand order repaired and checked by reading, not compiled, not run on hardware
//
#include <hip/hip_runtime.h>

// Problem constants (from reference)
#define B_  4
#define S_  2048
#define D_  1024
#define H_  16
#define DK_ 64

typedef __bf16 bf16;
typedef __attribute__((ext_vector_type(16))) __bf16 v16b;
typedef __attribute__((ext_vector_type(8)))  __bf16 v8b;
typedef __attribute__((ext_vector_type(4)))  __bf16 v4b;
typedef __attribute__((ext_vector_type(8)))  float  v8f;
typedef __attribute__((ext_vector_type(4)))  float  v4f;
typedef __attribute__((ext_vector_type(8)))  unsigned v8u;

__device__ __forceinline__ bf16 f2bf(float f) {
  union { float f; unsigned u; } v; v.f = f;
  unsigned r = v.u + 0x7fffu + ((v.u >> 16) & 1u);   // round-to-nearest-even
  unsigned short h = (unsigned short)(r >> 16);
  bf16 o; __builtin_memcpy(&o, &h, 2); return o;
}

__device__ __forceinline__ v8f wmma_bf16(v16b a, v16b b, v8f c) {
  return __builtin_amdgcn_wmma_f32_16x16x32_bf16(
      /*neg_a=*/false, a, /*neg_b=*/false, b,
      /*c_mod=*/(short)0, c, /*reuse_a=*/false, /*reuse_b=*/false);
}

// One DPP butterfly max stage (all lanes active).
template <int CTRL>
__device__ __forceinline__ float maxdpp(float x) {
  int y = __builtin_amdgcn_update_dpp(0, __float_as_int(x), CTRL, 0xf, 0xf, true);
  return fmaxf(x, __int_as_float(y));
}
// Max across the 16 lanes of each half-wave: xor1, xor2, mirror8, mirror16.
__device__ __forceinline__ float max16(float x) {
  x = maxdpp<0xB1>(x);    // quad_perm [1,0,3,2]  (lane ^ 1)
  x = maxdpp<0x4E>(x);    // quad_perm [2,3,0,1]  (lane ^ 2)
  x = maxdpp<0x141>(x);   // row_half_mirror      (lane ^ 7)
  x = maxdpp<0x140>(x);   // row_mirror           (lane ^ 15)
  return x;
}

// A-fragment (16x32, M x K), row-major source with K contiguous, row stride ld.
__device__ __forceinline__ v16b ldA(const bf16* base, int row0, int ld, int k0, int lane) {
  const int half = lane >> 4;
  const bf16* p = base + (size_t)(row0 + (lane & 15)) * ld + k0 + (half << 3);
  v8b lo = *(const v8b*)p;
  v8b hi = *(const v8b*)(p + 16);
  v16b f;
#pragma unroll
  for (int i = 0; i < 8; i++) { f[i] = lo[i]; f[i + 8] = hi[i]; }
  return f;
}

// B-fragment (32x16, K x N), source rows indexed by N with K contiguous (stride ld).
__device__ __forceinline__ v16b ldB(const bf16* base, int row0, int ld, int k0, int lane) {
  const bf16* p = base + (size_t)(row0 + (lane & 15)) * ld + k0 + ((lane >> 4) << 3);
  v8b lo = *(const v8b*)p;
  v8b hi = *(const v8b*)(p + 16);
  v16b f;
#pragma unroll
  for (int i = 0; i < 8; i++) { f[i] = lo[i]; f[i + 8] = hi[i]; }
  return f;
}

// ---------------- fp32 -> bf16 conversion ----------------
__global__ void cvt_f32_bf16(const float* __restrict__ in, bf16* __restrict__ out, int n4) {
  int i = blockIdx.x * blockDim.x + threadIdx.x;
  if (i >= n4) return;
  v4f x = *(const v4f*)(in + 4 * (size_t)i);
  v4b y;
#pragma unroll
  for (int j = 0; j < 4; j++) y[j] = f2bf(x[j]);
  *(v4b*)(out + 4 * (size_t)i) = y;
}

// ---------------- bf16 GEMM: Y = X @ W^T + bias, 64x64 tile per wave ----------------
// MODE 0: out bf16 [B,H,S,DK]; MODE 1: out bf16 [B,H,DK,S]; MODE 2: out f32 [M,N]
template <int MODE>
__global__ void __launch_bounds__(256)
gemm16(const bf16* __restrict__ X, const bf16* __restrict__ W,
       const float* __restrict__ bias, void* __restrict__ outp,
       int M, int N, int K) {
  const int lane = threadIdx.x & 31;
  const int wid  = blockIdx.x * (blockDim.x >> 5) + (threadIdx.x >> 5);
  const int nBlk = N >> 6;                // 64-col blocks
  const int m0 = (wid / nBlk) * 64;
  const int n0 = (wid % nBlk) * 64;
  if (m0 >= M) return;

  v8f acc[4][4];
#pragma unroll
  for (int mt = 0; mt < 4; mt++)
#pragma unroll
    for (int nt = 0; nt < 4; nt++)
#pragma unroll
      for (int r = 0; r < 8; r++) acc[mt][nt][r] = 0.0f;

  for (int k0 = 0; k0 < K; k0 += 32) {
    v16b bfr[4], afr[4];
#pragma unroll
    for (int nt = 0; nt < 4; nt++) bfr[nt] = ldB(W, n0 + 16 * nt, K, k0, lane);
#pragma unroll
    for (int mt = 0; mt < 4; mt++) afr[mt] = ldA(X, m0 + 16 * mt, K, k0, lane);
#pragma unroll
    for (int mt = 0; mt < 4; mt++)
#pragma unroll
      for (int nt = 0; nt < 4; nt++)
        acc[mt][nt] = wmma_bf16(afr[mt], bfr[nt], acc[mt][nt]);
  }

  const int half = lane >> 4, nl = lane & 15;

#pragma unroll
  for (int nt = 0; nt < 4; nt++) {
    const int n = n0 + 16 * nt + nl;
    const float bval = bias[n];
    const int h = n / DK_, dk = n % DK_;
#pragma unroll
    for (int mt = 0; mt < 4; mt++) {
      const int mrow0 = m0 + mt * 16 + half * 8;
      if (MODE == 2) {
        float* O = (float*)outp;
#pragma unroll
        for (int r = 0; r < 8; r++)
          O[(size_t)(mrow0 + r) * N + n] = acc[mt][nt][r] + bval;
      } else if (MODE == 0) {
        bf16* O = (bf16*)outp;
        const int b = mrow0 / S_, s0 = mrow0 % S_;
#pragma unroll
        for (int r = 0; r < 8; r++)
          O[(((size_t)(b * H_ + h)) * S_ + (s0 + r)) * DK_ + dk] = f2bf(acc[mt][nt][r] + bval);
      } else {  // MODE 1: transposed V, packed 16B store
        bf16* O = (bf16*)outp;
        const int b = mrow0 / S_, s0 = mrow0 % S_;
        v8b pk;
#pragma unroll
        for (int r = 0; r < 8; r++) pk[r] = f2bf(acc[mt][nt][r] + bval);
        *(v8b*)(O + (((size_t)(b * H_ + h)) * DK_ + dk) * S_ + s0) = pk;
      }
    }
  }
}

// ---------------- Flash attention: two 16-row q tiles (32 rows) per wave ----------------
// Q,K: bf16 [B,H,S,DK]; Vt: bf16 [B,H,DK,S]; out: bf16 [B,S,D]
__global__ void __launch_bounds__(256)
attn16(const bf16* __restrict__ Q, const bf16* __restrict__ Kh,
       const bf16* __restrict__ Vt, bf16* __restrict__ Ao) {
  __shared__ bf16 Pb[8][2][16 * 40];   // per-wave, per-qtile 16x32 P tile (pitch 40)
  const int lane = threadIdx.x & 31;
  const int w = threadIdx.x >> 5;
  const int wid = blockIdx.x * 8 + w;
  const int qBlocks = S_ / 32;
  const int qt = wid % qBlocks;        // 32-row q block
  const int bh = wid / qBlocks;
  const int b = bh / H_, h = bh % H_;

  const bf16* qH = Q  + (size_t)bh * S_ * DK_;
  const bf16* kH = Kh + (size_t)bh * S_ * DK_;
  const bf16* vH = Vt + (size_t)bh * DK_ * S_;

  v16b aq[2][2];
#pragma unroll
  for (int qi = 0; qi < 2; qi++) {
    aq[qi][0] = ldA(qH, qt * 32 + qi * 16, DK_, 0,  lane);
    aq[qi][1] = ldA(qH, qt * 32 + qi * 16, DK_, 32, lane);
  }

  v8u onesu;
#pragma unroll
  for (int i = 0; i < 8; i++) onesu[i] = 0x3F803F80u;   // packed bf16 {1.0, 1.0}
  const v16b ones = __builtin_bit_cast(v16b, onesu);

  v8f o[2][4], os[2];
  float rmax[2][8];
#pragma unroll
  for (int qi = 0; qi < 2; qi++) {
#pragma unroll
    for (int t = 0; t < 4; t++)
#pragma unroll
      for (int r = 0; r < 8; r++) o[qi][t][r] = 0.0f;
#pragma unroll
    for (int r = 0; r < 8; r++) { os[qi][r] = 0.0f; rmax[qi][r] = -3.0e38f; }
  }

  const int half = lane >> 4, nl = lane & 15;

  for (int kv = 0; kv < S_; kv += 32) {
    v16b kf[4];
    kf[0] = ldB(kH, kv,      DK_, 0,  lane);
    kf[1] = ldB(kH, kv,      DK_, 32, lane);
    kf[2] = ldB(kH, kv + 16, DK_, 0,  lane);
    kf[3] = ldB(kH, kv + 16, DK_, 32, lane);

#pragma unroll
    for (int qi = 0; qi < 2; qi++) {
      v8f c0, c1;
#pragma unroll
      for (int r = 0; r < 8; r++) { c0[r] = 0.0f; c1[r] = 0.0f; }
      c0 = wmma_bf16(aq[qi][0], kf[0], c0);
      c0 = wmma_bf16(aq[qi][1], kf[1], c0);
      c1 = wmma_bf16(aq[qi][0], kf[2], c1);
      c1 = wmma_bf16(aq[qi][1], kf[3], c1);

      bf16* myP = Pb[w][qi];
      float scale[8];
#pragma unroll
      for (int r = 0; r < 8; r++) {
        float x0 = c0[r] * 0.125f, x1 = c1[r] * 0.125f;   // 1/sqrt(DK)
        float mx = max16(fmaxf(x0, x1));                   // DPP row max
        float nm = fmaxf(rmax[qi][r], mx);
        float sf = __expf(rmax[qi][r] - nm);
        float p0 = __expf(x0 - nm), p1 = __expf(x1 - nm);
        rmax[qi][r] = nm;
        scale[r] = sf;
        const int m = r + half * 8;
        myP[m * 40 + nl]      = f2bf(p0);
        myP[m * 40 + 16 + nl] = f2bf(p1);
      }
#pragma unroll
      for (int t = 0; t < 4; t++)
#pragma unroll
        for (int r = 0; r < 8; r++) o[qi][t][r] *= scale[r];
#pragma unroll
      for (int r = 0; r < 8; r++) os[qi][r] *= scale[r];
    }

    asm volatile("s_wait_dscnt 0x0" ::: "memory");

    v16b vf0 = ldB(vH, 0,  S_, kv, lane);
    v16b vf1 = ldB(vH, 16, S_, kv, lane);
    v16b vf2 = ldB(vH, 32, S_, kv, lane);
    v16b vf3 = ldB(vH, 48, S_, kv, lane);

#pragma unroll
    for (int qi = 0; qi < 2; qi++) {
      v16b pf = ldA(Pb[w][qi], 0, 40, 0, lane);
      o[qi][0] = wmma_bf16(pf, vf0, o[qi][0]);
      o[qi][1] = wmma_bf16(pf, vf1, o[qi][1]);
      o[qi][2] = wmma_bf16(pf, vf2, o[qi][2]);
      o[qi][3] = wmma_bf16(pf, vf3, o[qi][3]);
      os[qi]   = wmma_bf16(pf, ones, os[qi]);   // row-sum of P via WMMA
    }
    asm volatile("s_wait_dscnt 0x0" ::: "memory");
  }

#pragma unroll
  for (int qi = 0; qi < 2; qi++) {
#pragma unroll
    for (int t = 0; t < 4; t++) {
#pragma unroll
      for (int r = 0; r < 8; r++) {
        float val = o[qi][t][r] / os[qi][r];
        const int s  = qt * 32 + qi * 16 + r + half * 8;
        const int dk = t * 16 + nl;
        Ao[(size_t)(b * S_ + s) * D_ + h * DK_ + dk] = f2bf(val);
      }
    }
  }
}

extern "C" void kernel_launch(void* const* d_in, const int* in_sizes, int n_in,
                              void* d_out, int out_size, void* d_ws, size_t ws_size,
                              hipStream_t stream) {
  (void)in_sizes; (void)n_in; (void)out_size; (void)ws_size;
  const float* q  = (const float*)d_in[0];
  const float* k  = (const float*)d_in[1];
  const float* v  = (const float*)d_in[2];
  const float* Wq = (const float*)d_in[3];
  const float* bq = (const float*)d_in[4];
  const float* Wk = (const float*)d_in[5];
  const float* bk = (const float*)d_in[6];
  const float* Wv = (const float*)d_in[7];
  const float* bv = (const float*)d_in[8];
  const float* Wo = (const float*)d_in[9];
  const float* bo = (const float*)d_in[10];

  char* ws = (char*)d_ws;
  size_t off = 0;
  auto carve = [&](size_t elems) -> bf16* {
    bf16* p = (bf16*)(ws + off);
    off += ((elems * sizeof(bf16)) + 255) & ~(size_t)255;
    return p;
  };
  const size_t BSD = (size_t)B_ * S_ * D_;
  const size_t DD  = (size_t)D_ * D_;
  bf16 *xq = carve(BSD), *xk = carve(BSD), *xv = carve(BSD);
  bf16 *wqb = carve(DD), *wkb = carve(DD), *wvb = carve(DD), *wob = carve(DD);
  bf16 *qb = carve(BSD), *kb = carve(BSD), *vt = carve(BSD), *ao = carve(BSD);

  auto cvt = [&](const float* src, bf16* dst, size_t n) {
    int n4 = (int)(n / 4);
    cvt_f32_bf16<<<(n4 + 255) / 256, 256, 0, stream>>>(src, dst, n4);
  };
  cvt(q, xq, BSD);  cvt(k, xk, BSD);  cvt(v, xv, BSD);
  cvt(Wq, wqb, DD); cvt(Wk, wkb, DD); cvt(Wv, wvb, DD); cvt(Wo, wob, DD);

  const int M = B_ * S_, N = D_, K = D_;
  const int gemmWaves = (M / 64) * (N / 64);           // 2048 waves
  dim3 gg(gemmWaves / 8), blk(256);
  gemm16<0><<<gg, blk, 0, stream>>>(xq, wqb, bq, qb, M, N, K);
  gemm16<0><<<gg, blk, 0, stream>>>(xk, wkb, bk, kb, M, N, K);
  gemm16<1><<<gg, blk, 0, stream>>>(xv, wvb, bv, vt, M, N, K);

  const int attnWaves = B_ * H_ * (S_ / 32);           // 4096 waves
  attn16<<<attnWaves / 8, 256, 0, stream>>>(qb, kb, vt, ao);

  gemm16<2><<<gg, blk, 0, stream>>>(ao, wob, bo, d_out, M, N, K);
}
